// MultiHeadAttention_25151328485977
// MI455X (gfx1250) — compile-verified, matrix operand order checked by reading, not run on hardware
//
#include <hip/hip_runtime.h>
#include <hip/hip_bf16.h>

// ---------------------------------------------------------------------------
// MI455X (gfx1250) multi-head causal attention, bf16 WMMA pipeline.
// B=2, T=2048, E=768, H=12, Dh=64.
// v3: attention uses GLOBAL_LOAD_ASYNC_TO_LDS_B128 double-buffered K/V blocks
//     (ASYNCcnt-gated) and 32 queries/wave; GEMMs stay 64x64 double-buffered.
// ---------------------------------------------------------------------------

typedef __bf16 bf16_t;
typedef __attribute__((ext_vector_type(16))) __bf16 v16bf;
typedef __attribute__((ext_vector_type(8)))  __bf16 v8bf;
typedef __attribute__((ext_vector_type(8)))  float  v8f;

#define EMBED    768
#define HEADS    12
#define HEAD_DIM 64
#define BATCH    2
#define SEQ      2048
#define ROWS     (BATCH * SEQ)          // 4096

// ---------------------------------------------------------------------------
// Fragment loader: 16x32 bf16 tile from a row-major matrix (global or LDS).
// Layout per ISA 7.12.2. Two contiguous 16B loads per lane.
// ---------------------------------------------------------------------------
__device__ __forceinline__ v16bf load_frag16x32(const bf16_t* __restrict__ base,
                                                int ld, int lane) {
  const int r  = lane & 15;            // row (A) / column (B) index
  const int kb = (lane >> 4) << 3;     // K base: 0 or 8
  const bf16_t* p = base + r * ld + kb;
  v8bf lo = *reinterpret_cast<const v8bf*>(p);        // K = kb .. kb+7
  v8bf hi = *reinterpret_cast<const v8bf*>(p + 16);   // K = kb+16 .. kb+23
  return __builtin_shufflevector(lo, hi, 0, 1, 2, 3, 4, 5, 6, 7,
                                         8, 9, 10, 11, 12, 13, 14, 15);
}

__device__ __forceinline__ v8f wmma_bf16(v16bf a, v16bf b, v8f c) {
  return __builtin_amdgcn_wmma_f32_16x16x32_bf16(false, a, false, b,
                                                 (short)0, c, false, false);
}

// One async 512B tile move (32 lanes x 16B) global -> LDS, ASYNCcnt-tracked.
__device__ __forceinline__ void async_b128(unsigned int lds_off, const void* g) {
  asm volatile("global_load_async_to_lds_b128 %0, %1, off"
               :: "v"(lds_off), "v"(g) : "memory");
}

// Load one 64xK-half slab: 4 A fragments (rows) + 4 B fragments (cols),
// both from ld=768 row-major storage at K offset k0.
__device__ __forceinline__ void load_ab64(const bf16_t* __restrict__ A,
                                          const bf16_t* __restrict__ Bt,
                                          int k0, int lane,
                                          v16bf a[4], v16bf b[4]) {
#pragma unroll
  for (int i = 0; i < 4; ++i)
    a[i] = load_frag16x32(A + (size_t)(i * 16) * EMBED + k0, EMBED, lane);
#pragma unroll
  for (int i = 0; i < 4; ++i)
    b[i] = load_frag16x32(Bt + (size_t)(i * 16) * EMBED + k0, EMBED, lane);
}

// ---------------------------------------------------------------------------
// Prep: fp32 -> bf16 conversions (+ weight transposes so all WMMA B-fragment
// loads are contiguous).
// ---------------------------------------------------------------------------
__global__ void cvt_x_kernel(const float* __restrict__ x,
                             bf16_t* __restrict__ xb, int n) {
  int i = blockIdx.x * blockDim.x + threadIdx.x;
  if (i < n) xb[i] = (bf16_t)x[i];
}

// wqkvt[((z*H + h)*64 + d)*768 + e] = W_z[h][e][d]
__global__ void cvt_wqkv_kernel(const float* __restrict__ Wq,
                                const float* __restrict__ Wk,
                                const float* __restrict__ Wv,
                                bf16_t* __restrict__ wt) {
  int i = blockIdx.x * blockDim.x + threadIdx.x;
  const int total = 3 * HEADS * HEAD_DIM * EMBED;
  if (i >= total) return;
  int e = i % EMBED;
  int d = (i / EMBED) % HEAD_DIM;
  int h = (i / (EMBED * HEAD_DIM)) % HEADS;
  int z = i / (EMBED * HEAD_DIM * HEADS);
  const float* W = (z == 0) ? Wq : (z == 1) ? Wk : Wv;
  wt[i] = (bf16_t)W[((size_t)h * EMBED + e) * HEAD_DIM + d];
}

// wpt[f*768 + e] = Wp[e][f]
__global__ void cvt_wp_kernel(const float* __restrict__ Wp,
                              bf16_t* __restrict__ wpt) {
  int i = blockIdx.x * blockDim.x + threadIdx.x;
  if (i >= EMBED * EMBED) return;
  int e = i % EMBED;
  int f = i / EMBED;
  wpt[i] = (bf16_t)Wp[(size_t)e * EMBED + f];
}

// ---------------------------------------------------------------------------
// QKV projection: per (head,z): [4096 x 768] @ [768 x 64]. One wave owns a
// 64x64 output tile; 64-wide K steps, double-buffered fragments.
// q,k stored [B,H,T,64]; v stored transposed [B,H,64,T].
// ---------------------------------------------------------------------------
__global__ void __launch_bounds__(128)
qkv_kernel(const bf16_t* __restrict__ xb, const bf16_t* __restrict__ wqkvt,
           bf16_t* __restrict__ q, bf16_t* __restrict__ k,
           bf16_t* __restrict__ vT) {
  const int lane   = threadIdx.x & 31;
  const int wid    = threadIdx.x >> 5;
  const int mTile  = blockIdx.x * 4 + wid;    // 0..63 (64-row tiles)
  const int h      = blockIdx.y;              // 0..11
  const int z      = blockIdx.z;              // 0=q 1=k 2=v
  const int row0   = mTile * 64;

  const bf16_t* A  = xb + (size_t)row0 * EMBED;
  const bf16_t* Bt = wqkvt + (size_t)(z * HEADS + h) * HEAD_DIM * EMBED;

  v8f acc[4][4] = {};
  v16bf a0[4], b0[4], a1[4], b1[4];
  load_ab64(A, Bt, 0, lane, a0, b0);

  for (int k0 = 0; k0 < EMBED; k0 += 64) {
    load_ab64(A, Bt, k0 + 32, lane, a1, b1);          // prefetch half 2
#pragma unroll
    for (int nt = 0; nt < 4; ++nt)
#pragma unroll
      for (int mt = 0; mt < 4; ++mt)
        acc[mt][nt] = wmma_bf16(a0[mt], b0[nt], acc[mt][nt]);
    const int kn = (k0 + 64 < EMBED) ? k0 + 64 : EMBED - 64;  // clamp (dup ok)
    load_ab64(A, Bt, kn, lane, a0, b0);               // prefetch next half 1
#pragma unroll
    for (int nt = 0; nt < 4; ++nt)
#pragma unroll
      for (int mt = 0; mt < 4; ++mt)
        acc[mt][nt] = wmma_bf16(a1[mt], b1[nt], acc[mt][nt]);
  }

  const int col   = lane & 15;
  const int rHalf = (lane >> 4) << 3;
  const int bIdx  = row0 >> 11;
  const int tBase = row0 & (SEQ - 1);
  const size_t bh = (size_t)bIdx * HEADS + h;

#pragma unroll
  for (int mt = 0; mt < 4; ++mt)
#pragma unroll
    for (int nt = 0; nt < 4; ++nt)
#pragma unroll
      for (int r = 0; r < 8; ++r) {
        int t = tBase + mt * 16 + rHalf + r;
        int d = nt * 16 + col;
        bf16_t val = (bf16_t)acc[mt][nt][r];
        if (z == 2)      vT[(bh * HEAD_DIM + d) * SEQ + t] = val;
        else if (z == 1) k [(bh * SEQ + t) * HEAD_DIM + d] = val;
        else             q [(bh * SEQ + t) * HEAD_DIM + d] = val;
      }
}

// ---------------------------------------------------------------------------
// Flash attention: one wave (one workgroup) = one 32-query tile of one (b,h).
// K/V staged into LDS by the async-copy engine, double-buffered, gated with
// s_wait_asynccnt. Per 64-key block: 32 WMMA, all operands from LDS.
//   K LDS block: [64 keys][64 d]  (row-major, 8 KB)
//   V LDS block: [64 d][64 keys]  (row-major, 8 KB)  from transposed vT
// ---------------------------------------------------------------------------
__global__ void __launch_bounds__(32)
attn_kernel(const bf16_t* __restrict__ q, const bf16_t* __restrict__ k,
            const bf16_t* __restrict__ vT, bf16_t* __restrict__ attn) {
  __shared__ __align__(16) bf16_t ldsK[2][64 * 64];   // 16 KB
  __shared__ __align__(16) bf16_t ldsV[2][64 * 64];   // 16 KB
  __shared__ __align__(16) bf16_t plds[16 * 64];      //  2 KB

  const int lane = threadIdx.x & 31;
  const int gw   = blockIdx.x;              // 0..1535
  const int qt   = gw & 63;                 // 32-query tile within (b,h)
  const int bh   = gw >> 6;                 // 0..23
  const int t0   = qt * 32;

  const bf16_t* Q = q  + ((size_t)bh * SEQ + t0) * HEAD_DIM;
  const bf16_t* K = k  + (size_t)bh * SEQ * HEAD_DIM;
  const bf16_t* V = vT + (size_t)bh * HEAD_DIM * SEQ;

  const unsigned int kOff[2] = {(unsigned int)(uintptr_t)&ldsK[0][0],
                                (unsigned int)(uintptr_t)&ldsK[1][0]};
  const unsigned int vOff[2] = {(unsigned int)(uintptr_t)&ldsV[0][0],
                                (unsigned int)(uintptr_t)&ldsV[1][0]};

  const int col   = lane & 15;
  const int rHalf = (lane >> 4) << 3;

  // Q fragments: 2 query subtiles x 2 K-halves, reused for every key block.
  v16bf aq[2][2];
#pragma unroll
  for (int qh = 0; qh < 2; ++qh) {
    aq[qh][0] = load_frag16x32(Q + (size_t)(qh * 16) * HEAD_DIM + 0,  HEAD_DIM, lane);
    aq[qh][1] = load_frag16x32(Q + (size_t)(qh * 16) * HEAD_DIM + 32, HEAD_DIM, lane);
  }

  float m[2][8], l[2][8];
  v8f o[2][4] = {};
#pragma unroll
  for (int qh = 0; qh < 2; ++qh)
#pragma unroll
    for (int r = 0; r < 8; ++r) { m[qh][r] = -3.0e38f; l[qh][r] = 0.0f; }

  const float scale = 0.125f;   // 1/sqrt(64)
  const int sEnd = t0 + 31;

  // ---- async stage of one 64-key K/V block into LDS buffer `buf` ---------
  auto stage_block = [&](int s0, int buf) {
    // K: 64 keys x 128B contiguous. chunk = i*32+lane -> 16B each.
    const char* kg = (const char*)K + (size_t)s0 * (HEAD_DIM * 2) + lane * 16;
#pragma unroll
    for (int i = 0; i < 16; ++i)
      async_b128(kOff[buf] + lane * 16 + i * 512, kg + i * 512);
    // V: 64 rows (d), 128B slice per row, row stride SEQ*2 bytes.
    // chunk = i*32+lane: row = i*4 + (lane>>3), byteInRow = (lane&7)*16.
    const char* vg = (const char*)V + (size_t)s0 * 2 +
                     (size_t)(lane >> 3) * (SEQ * 2) + (lane & 7) * 16;
#pragma unroll
    for (int i = 0; i < 16; ++i)
      async_b128(vOff[buf] + lane * 16 + i * 512, vg + (size_t)i * 4 * (SEQ * 2));
  };

  stage_block(0, 0);

  for (int s0 = 0; s0 <= sEnd; s0 += 64) {
    const int cur = (s0 >> 6) & 1;
    if (s0 + 64 <= sEnd) {
      stage_block(s0 + 64, cur ^ 1);                    // 32 async in flight
      asm volatile("s_wait_asynccnt 0x20" ::: "memory"); // current block landed
    } else {
      asm volatile("s_wait_asynccnt 0x0" ::: "memory");
    }
    const bf16_t* lk = ldsK[cur];
    const bf16_t* lv = ldsV[cur];

#pragma unroll
    for (int qh = 0; qh < 2; ++qh) {
      const int tq0 = t0 + qh * 16;

      // ---- scores: four 16x16 subtiles over 64 keys (LDS operands) -------
      v8f S[4] = {};
#pragma unroll
      for (int sub = 0; sub < 4; ++sub) {
        v16bf bk0 = load_frag16x32(lk + (sub * 16) * 64 + 0,  64, lane);
        v16bf bk1 = load_frag16x32(lk + (sub * 16) * 64 + 32, 64, lane);
        S[sub] = wmma_bf16(aq[qh][0], bk0, S[sub]);
        S[sub] = wmma_bf16(aq[qh][1], bk1, S[sub]);
      }

      // ---- scale (+ causal mask only near the diagonal) ------------------
      if (s0 + 63 > tq0) {
#pragma unroll
        for (int sub = 0; sub < 4; ++sub) {
          int s = s0 + sub * 16 + col;
#pragma unroll
          for (int r = 0; r < 8; ++r) {
            int t = tq0 + rHalf + r;
            float v = S[sub][r] * scale;
            S[sub][r] = (s <= t) ? v : -3.0e38f;
          }
        }
      } else {
#pragma unroll
        for (int sub = 0; sub < 4; ++sub)
#pragma unroll
          for (int r = 0; r < 8; ++r)
            S[sub][r] *= scale;
      }

      // ---- online softmax update (once per 64 keys) ----------------------
#pragma unroll
      for (int r = 0; r < 8; ++r) {
        float rm = fmaxf(fmaxf(S[0][r], S[1][r]), fmaxf(S[2][r], S[3][r]));
#pragma unroll
        for (int mk = 1; mk < 16; mk <<= 1)
          rm = fmaxf(rm, __shfl_xor(rm, mk, 32));
        float mnew  = fmaxf(m[qh][r], rm);
        float alpha = __expf(m[qh][r] - mnew);
        float p0 = __expf(S[0][r] - mnew);
        float p1 = __expf(S[1][r] - mnew);
        float p2 = __expf(S[2][r] - mnew);
        float p3 = __expf(S[3][r] - mnew);
        S[0][r] = p0; S[1][r] = p1; S[2][r] = p2; S[3][r] = p3;
        float rs = (p0 + p1) + (p2 + p3);
#pragma unroll
        for (int mk = 1; mk < 16; mk <<= 1)
          rs += __shfl_xor(rs, mk, 32);
        l[qh][r] = l[qh][r] * alpha + rs;
        m[qh][r] = mnew;
#pragma unroll
        for (int d = 0; d < 4; ++d) o[qh][d][r] *= alpha;
      }

      // ---- re-layout P (C-frag -> A-frag) via LDS, 16x64 tile ------------
#pragma unroll
      for (int sub = 0; sub < 4; ++sub)
#pragma unroll
        for (int r = 0; r < 8; ++r)
          plds[(rHalf + r) * 64 + sub * 16 + col] = (bf16_t)S[sub][r];
      asm volatile("s_wait_dscnt 0" ::: "memory");
      v16bf pa0 = load_frag16x32(plds + 0,  64, lane);
      v16bf pa1 = load_frag16x32(plds + 32, 64, lane);

      // ---- O += P @ V (K = 64 -> 2 WMMA per d-tile, LDS operands) --------
#pragma unroll
      for (int d = 0; d < 4; ++d) {
        v16bf bv0 = load_frag16x32(lv + (d * 16) * 64 + 0,  64, lane);
        v16bf bv1 = load_frag16x32(lv + (d * 16) * 64 + 32, 64, lane);
        o[qh][d] = wmma_bf16(pa0, bv0, o[qh][d]);
        o[qh][d] = wmma_bf16(pa1, bv1, o[qh][d]);
      }
      asm volatile("s_wait_dscnt 0" ::: "memory");  // P-tile reuse across qh
    }
  }

  // ---- normalize and store to concat-heads layout [4096 x 768] ----------
  const int bIdx = bh / HEADS, h = bh % HEADS;
#pragma unroll
  for (int qh = 0; qh < 2; ++qh) {
#pragma unroll
    for (int r = 0; r < 8; ++r) {
      float inv = 1.0f / l[qh][r];
#pragma unroll
      for (int d = 0; d < 4; ++d) o[qh][d][r] *= inv;
    }
#pragma unroll
    for (int d = 0; d < 4; ++d)
#pragma unroll
      for (int r = 0; r < 8; ++r) {
        int t = t0 + qh * 16 + rHalf + r;
        attn[((size_t)(bIdx * SEQ + t)) * EMBED + h * HEAD_DIM + d * 16 + col] =
            (bf16_t)o[qh][d][r];
      }
  }
}

// ---------------------------------------------------------------------------
// Output projection: [4096 x 768] @ Wp[768 x 768] + bp -> fp32 out.
// ---------------------------------------------------------------------------
__global__ void __launch_bounds__(128)
proj_kernel(const bf16_t* __restrict__ attn, const bf16_t* __restrict__ wpt,
            const float* __restrict__ bp, float* __restrict__ out) {
  const int lane  = threadIdx.x & 31;
  const int wid   = threadIdx.x >> 5;
  const int mTile = blockIdx.x * 4 + wid;     // 0..63
  const int n0    = blockIdx.y * 64;          // 0..704
  const int row0  = mTile * 64;

  const bf16_t* A  = attn + (size_t)row0 * EMBED;
  const bf16_t* Bt = wpt + (size_t)n0 * EMBED;

  v8f acc[4][4] = {};
  v16bf a0[4], b0[4], a1[4], b1[4];
  load_ab64(A, Bt, 0, lane, a0, b0);

  for (int k0 = 0; k0 < EMBED; k0 += 64) {
    load_ab64(A, Bt, k0 + 32, lane, a1, b1);
#pragma unroll
    for (int nt = 0; nt < 4; ++nt)
#pragma unroll
      for (int mt = 0; mt < 4; ++mt)
        acc[mt][nt] = wmma_bf16(a0[mt], b0[nt], acc[mt][nt]);
    const int kn = (k0 + 64 < EMBED) ? k0 + 64 : EMBED - 64;
    load_ab64(A, Bt, kn, lane, a0, b0);
#pragma unroll
    for (int nt = 0; nt < 4; ++nt)
#pragma unroll
      for (int mt = 0; mt < 4; ++mt)
        acc[mt][nt] = wmma_bf16(a1[mt], b1[nt], acc[mt][nt]);
  }

  const int col   = lane & 15;
  const int rHalf = (lane >> 4) << 3;
#pragma unroll
  for (int nt = 0; nt < 4; ++nt) {
    int n = n0 + nt * 16 + col;
    float bias = bp[n];
#pragma unroll
    for (int mt = 0; mt < 4; ++mt)
#pragma unroll
      for (int r = 0; r < 8; ++r)
        out[(size_t)(row0 + mt * 16 + rHalf + r) * EMBED + n] =
            acc[mt][nt][r] + bias;
  }
}

// ---------------------------------------------------------------------------
// Launch. Workspace layout (bytes, all 256-aligned):
//   xb     @ 0         : 6,291,456
//   wqkvt  @ 6291456   : 3,538,944
//   wpt    @ 9830400   : 1,179,648
//   q      @ 11010048  : 6,291,456
//   k      @ 17301504  : 6,291,456
//   vT     @ 23592960  : 6,291,456
//   attn   @ 29884416  : 6,291,456   (ends 36,175,872)
// ---------------------------------------------------------------------------
extern "C" void kernel_launch(void* const* d_in, const int* in_sizes, int n_in,
                              void* d_out, int out_size, void* d_ws,
                              size_t ws_size, hipStream_t stream) {
  (void)in_sizes; (void)n_in; (void)out_size; (void)ws_size;
  const float* x  = (const float*)d_in[0];
  const float* Wq = (const float*)d_in[1];
  const float* Wk = (const float*)d_in[2];
  const float* Wv = (const float*)d_in[3];
  const float* Wp = (const float*)d_in[4];
  const float* bp = (const float*)d_in[5];
  float* out = (float*)d_out;

  char* ws = (char*)d_ws;
  bf16_t* xb    = (bf16_t*)(ws + 0);
  bf16_t* wqkvt = (bf16_t*)(ws + 6291456);
  bf16_t* wpt   = (bf16_t*)(ws + 9830400);
  bf16_t* qb    = (bf16_t*)(ws + 11010048);
  bf16_t* kb    = (bf16_t*)(ws + 17301504);
  bf16_t* vT    = (bf16_t*)(ws + 23592960);
  bf16_t* attn  = (bf16_t*)(ws + 29884416);

  const int nx = ROWS * EMBED;                      // 3,145,728
  const int nw = 3 * HEADS * HEAD_DIM * EMBED;      // 1,769,472
  const int np = EMBED * EMBED;                     //   589,824

  cvt_x_kernel   <<<(nx + 255) / 256, 256, 0, stream>>>(x, xb, nx);
  cvt_wqkv_kernel<<<(nw + 255) / 256, 256, 0, stream>>>(Wq, Wk, Wv, wqkvt);
  cvt_wp_kernel  <<<(np + 255) / 256, 256, 0, stream>>>(Wp, wpt);

  qkv_kernel <<<dim3(16, HEADS, 3), 128, 0, stream>>>(xb, wqkvt, qb, kb, vT);
  attn_kernel<<<1536, 32, 0, stream>>>(qb, kb, vT, attn);
  proj_kernel<<<dim3(16, HEADS), 128, 0, stream>>>(attn, wpt, bp, out);
}
